// SpatialTemporalAttention3D_41403484733894
// MI455X (gfx1250) — compile-verified, matrix operand order checked by reading, not run on hardware
//
#include <hip/hip_runtime.h>
#include <hip/hip_bf16.h>

#define C_DIM 512
#define N_TOK 4096
#define NHEAD 4
#define DHEAD 128
#define NGROUPS 32
#define CH_PER_G 16
#define EPS 1e-5f

typedef __attribute__((ext_vector_type(16))) __bf16 v16bf;
typedef __attribute__((ext_vector_type(8)))  float  v8f;
typedef __attribute__((ext_vector_type(8)))  unsigned short ushort8;
typedef __attribute__((ext_vector_type(4)))  unsigned short ushort4v;
typedef __attribute__((ext_vector_type(4)))  unsigned int   uint32x4;
typedef __attribute__((ext_vector_type(8)))  int            int32x8;
typedef __attribute__((ext_vector_type(4)))  int            int32x4;

union FragBF {
    ushort8 u[2];
    v16bf   b;
};

__device__ inline unsigned short f2bf(float f) {
    union { float f; unsigned int u; } cv; cv.f = f;
    unsigned int u = cv.u;
    u += 0x7fffu + ((u >> 16) & 1u);   // round-to-nearest-even
    return (unsigned short)(u >> 16);
}

// ---- Tensor Data Mover: 2D tile (bf16) global -> LDS, packed destination ----
// D# per CDNA5 ISA. Issue from ONE wave only.
__device__ inline void tdm_load_2d_bf16(unsigned int lds_off,
                                        unsigned long long gaddr,
                                        unsigned int tensor_d0, unsigned int tensor_d1,
                                        unsigned int tile_d0,   unsigned int tile_d1,
                                        unsigned long long stride0) {
    uint32x4 g0;
    g0[0] = 1u;                                        // count=1, user descriptor
    g0[1] = lds_off;                                   // LDS byte address
    g0[2] = (unsigned int)(gaddr & 0xffffffffu);       // global_addr[31:0]
    g0[3] = (unsigned int)((gaddr >> 32) & 0x1ffffffu) // global_addr[56:32]
            | (2u << 30);                              // type = 2 ("image")
    int32x8 g1;
    g1[0] = (int)(1u << 16);                           // data_size=1 -> 2 bytes
    g1[1] = (int)((tensor_d0 & 0xffffu) << 16);        // tensor_dim0[15:0]
    g1[2] = (int)((tensor_d0 >> 16) | ((tensor_d1 & 0xffffu) << 16));
    g1[3] = (int)((tensor_d1 >> 16) | (tile_d0 << 16));// tile_dim0
    g1[4] = (int)tile_d1;                              // tile_dim1 (tile_dim2=0)
    g1[5] = (int)(unsigned int)(stride0 & 0xffffffffu);// tensor_dim0_stride lo
    g1[6] = (int)(unsigned int)(stride0 >> 32);        // stride hi (dim1_stride=0)
    g1[7] = 0;
    int32x4 gz = {0, 0, 0, 0};
#if defined(__clang_major__) && (__clang_major__ >= 23)
    int32x8 gz8 = {0, 0, 0, 0, 0, 0, 0, 0};
    __builtin_amdgcn_tensor_load_to_lds(g0, g1, gz, gz, gz8, 0);
#else
    __builtin_amdgcn_tensor_load_to_lds(g0, g1, gz, gz, 0);
#endif
}

// ---------------- GroupNorm statistics: one block per group ----------------
__global__ __launch_bounds__(256) void gn_stats(const float* __restrict__ x,
                                                float* __restrict__ stats) {
    const int g = blockIdx.x;
    const int tid = threadIdx.x;
    const float* xg = x + (size_t)g * CH_PER_G * N_TOK;
    const int total = CH_PER_G * N_TOK;   // 65536
    float s = 0.f, ss = 0.f;
    for (int i = tid * 4; i < total; i += 256 * 4) {
        float4 v = *(const float4*)(xg + i);
        s  += v.x + v.y + v.z + v.w;
        ss += v.x * v.x + v.y * v.y + v.z * v.z + v.w * v.w;
    }
    __shared__ float red[512];
    red[tid] = s; red[256 + tid] = ss;
    __syncthreads();
    for (int st = 128; st > 0; st >>= 1) {
        if (tid < st) { red[tid] += red[tid + st]; red[256 + tid] += red[256 + tid + st]; }
        __syncthreads();
    }
    if (tid == 0) {
        float mean = red[0] / (float)total;
        float var  = red[256] / (float)total - mean * mean;
        stats[g]      = mean;
        stats[32 + g] = rsqrtf(var + EPS);
    }
}

// ---------------- Normalize + affine -> bf16 h ----------------
__global__ __launch_bounds__(256) void gn_apply(const float* __restrict__ x,
                                                const float* __restrict__ stats,
                                                const float* __restrict__ w,
                                                const float* __restrict__ b,
                                                unsigned short* __restrict__ h) {
    const int idx = (blockIdx.x * 256 + threadIdx.x) * 4;   // over C*N
    const int c = idx / N_TOK;
    const int g = c / CH_PER_G;
    const float mean = stats[g], rstd = stats[32 + g];
    const float sw = w[c] * rstd;
    const float sb = b[c] - mean * sw;
    float4 v = *(const float4*)(x + idx);
    ushort4v o;
    o[0] = f2bf(v.x * sw + sb);
    o[1] = f2bf(v.y * sw + sb);
    o[2] = f2bf(v.z * sw + sb);
    o[3] = f2bf(v.w * sw + sb);
    *(ushort4v*)(h + idx) = o;
}

// ---------------- Convert weights to bf16 ----------------
__global__ __launch_bounds__(256) void cvt_w(const float* __restrict__ qw,
                                             const float* __restrict__ pw,
                                             unsigned short* __restrict__ qwb,
                                             unsigned short* __restrict__ pwb) {
    const int i = blockIdx.x * 256 + threadIdx.x;
    if (i < 3 * C_DIM * C_DIM) qwb[i] = f2bf(qw[i]);
    if (i < C_DIM * C_DIM)     pwb[i] = f2bf(pw[i]);
}

// ---------------- Tiled WMMA GEMM: out[M,4096] = A[M,512] * B[512,4096] ----------------
// A tile streamed by TDM (tensor_load_to_lds); B tile transposed through VGPRs.
// MODE 0: +bias -> bf16 out (QKV).  MODE 1: +bias +residual -> f32 out (proj).
template <int MODE>
__global__ __launch_bounds__(256) void gemm_wmma(const unsigned short* __restrict__ A,
                                                 const unsigned short* __restrict__ B,
                                                 const float* __restrict__ bias,
                                                 const float* __restrict__ resid,
                                                 unsigned short* __restrict__ out_bf,
                                                 float* __restrict__ out_f) {
    __shared__ __align__(16) unsigned short As[128][32];  // [m][k]
    __shared__ __align__(16) unsigned short Bs[128][32];  // [n][k] (transposed in LDS)
    const int tid  = threadIdx.x;
    const int wave = tid >> 5;
    const int lane = tid & 31;
    const int hf   = lane >> 4;
    const int l16  = lane & 15;
    const int m0 = blockIdx.y * 128;
    const int n0 = blockIdx.x * 128;
    const unsigned int as_off = (unsigned int)(size_t)(void*)&As[0][0];

    v8f acc[8] = {};

    for (int k0 = 0; k0 < C_DIM; k0 += 32) {
        __syncthreads();
        // A tile 128x32 via Tensor Data Mover (wave 0 issues; EXEC-uniform branch)
        if (wave == 0) {
            tdm_load_2d_bf16(as_off,
                             (unsigned long long)(size_t)(A + (size_t)m0 * C_DIM + k0),
                             /*tensor_d0=*/C_DIM, /*tensor_d1=*/128,
                             /*tile_d0=*/32, /*tile_d1=*/128,
                             /*stride0=*/C_DIM);
        }
        // B tile 32x128 -> Bs[n][k] (transpose through LDS)
        {
            int kk = tid >> 4;
            int nn = (tid & 15) << 3;
            // prefetch next K-step's B rows into cache while we work
            if (k0 + 32 < C_DIM)
                __builtin_prefetch(B + (size_t)(k0 + 32 + kk) * N_TOK + n0 + nn, 0, 0);
            for (int cc = tid; cc < 512; cc += 256) {
                int k2 = cc >> 4;
                int n2 = (cc & 15) << 3;
                uint4 v = *(const uint4*)(B + (size_t)(k0 + k2) * N_TOK + n0 + n2);
                const unsigned short* pv = (const unsigned short*)&v;
#pragma unroll
                for (int j = 0; j < 8; ++j) Bs[n2 + j][k2] = pv[j];
            }
        }
        if (wave == 0) __builtin_amdgcn_s_wait_tensorcnt(0);
        __syncthreads();

        FragBF fa;
        fa.u[0] = *(const ushort8*)&As[wave * 16 + l16][hf * 8];
        fa.u[1] = *(const ushort8*)&As[wave * 16 + l16][16 + hf * 8];
#pragma unroll
        for (int j = 0; j < 8; ++j) {
            FragBF fb;
            fb.u[0] = *(const ushort8*)&Bs[j * 16 + l16][hf * 8];
            fb.u[1] = *(const ushort8*)&Bs[j * 16 + l16][16 + hf * 8];
            acc[j] = __builtin_amdgcn_wmma_f32_16x16x32_bf16(
                false, fa.b, false, fb.b, (short)0, acc[j], false, false);
        }
    }

#pragma unroll
    for (int j = 0; j < 8; ++j) {
#pragma unroll
        for (int r = 0; r < 8; ++r) {
            const int row = m0 + wave * 16 + r + 8 * hf;
            const int col = n0 + j * 16 + l16;
            const float v = acc[j][r] + bias[row];
            const size_t idx = (size_t)row * N_TOK + col;
            if (MODE == 0) out_bf[idx] = f2bf(v);
            else           out_f[idx]  = v + resid[idx];
        }
    }
}

// ---------------- Flash attention per head (d=128, N=4096) ----------------
// Block: 4 waves = 64 query columns; stream 64-wide K/V tiles.
// V tile (direct layout) streamed by TDM; Q/K tiles transposed through VGPRs.
__global__ __launch_bounds__(128) void flash_attn(const unsigned short* __restrict__ qkv,
                                                  unsigned short* __restrict__ out) {
    __shared__ __align__(16) unsigned short qs[64][128];   // [n][d]
    __shared__ __align__(16) unsigned short ks[64][128];   // [m][d]
    __shared__ __align__(16) unsigned short vs[128][64];   // [d][m]
    __shared__ __align__(16) unsigned short ps[64][64];    // [n][m]
    const int tid  = threadIdx.x;
    const int wave = tid >> 5;
    const int lane = tid & 31;
    const int hf   = lane >> 4;
    const int l16  = lane & 15;
    const int head  = blockIdx.y;
    const int nbase = blockIdx.x * 64;
    const unsigned short* Q = qkv + (size_t)(head * DHEAD) * N_TOK;
    const unsigned short* K = qkv + (size_t)(C_DIM + head * DHEAD) * N_TOK;
    const unsigned short* V = qkv + (size_t)(2 * C_DIM + head * DHEAD) * N_TOK;
    const float scale = 0.08838834764831845f;   // 1/sqrt(128)
    const unsigned int vs_off = (unsigned int)(size_t)(void*)&vs[0][0];

    // Q tile 128d x 64n, transpose into qs[n][d]
    for (int cc = tid; cc < 1024; cc += 128) {
        int dd = cc >> 3;
        int nn = (cc & 7) << 3;
        uint4 v = *(const uint4*)(Q + (size_t)dd * N_TOK + nbase + nn);
        const unsigned short* pv = (const unsigned short*)&v;
#pragma unroll
        for (int j = 0; j < 8; ++j) qs[nn + j][dd] = pv[j];
    }

    float Mrow[8], Lrow[8];
    v8f oacc[8] = {};
#pragma unroll
    for (int r = 0; r < 8; ++r) { Mrow[r] = -1e30f; Lrow[r] = 0.f; }
    __syncthreads();

    for (int mbase = 0; mbase < N_TOK; mbase += 64) {
        // V tile 128d x 64m via Tensor Data Mover into packed vs[d][m]
        if (wave == 0) {
            tdm_load_2d_bf16(vs_off,
                             (unsigned long long)(size_t)(V + mbase),
                             /*tensor_d0=*/N_TOK, /*tensor_d1=*/DHEAD,
                             /*tile_d0=*/64, /*tile_d1=*/DHEAD,
                             /*stride0=*/N_TOK);
        }
        // K tile 128d x 64m, transpose into ks[m][d]
        for (int cc = tid; cc < 1024; cc += 128) {
            int dd = cc >> 3;
            int nn = (cc & 7) << 3;
            uint4 v = *(const uint4*)(K + (size_t)dd * N_TOK + mbase + nn);
            const unsigned short* pv = (const unsigned short*)&v;
#pragma unroll
            for (int j = 0; j < 8; ++j) ks[nn + j][dd] = pv[j];
        }
        if (wave == 0) __builtin_amdgcn_s_wait_tensorcnt(0);
        __syncthreads();

        // S = Q^T K  (16n x 64m per wave), accumulate over d=128
        v8f sacc[4] = {};
#pragma unroll
        for (int kk = 0; kk < 4; ++kk) {
            FragBF fa;
            fa.u[0] = *(const ushort8*)&qs[wave * 16 + l16][kk * 32 + hf * 8];
            fa.u[1] = *(const ushort8*)&qs[wave * 16 + l16][kk * 32 + 16 + hf * 8];
#pragma unroll
            for (int t = 0; t < 4; ++t) {
                FragBF fb;
                fb.u[0] = *(const ushort8*)&ks[t * 16 + l16][kk * 32 + hf * 8];
                fb.u[1] = *(const ushort8*)&ks[t * 16 + l16][kk * 32 + 16 + hf * 8];
                sacc[t] = __builtin_amdgcn_wmma_f32_16x16x32_bf16(
                    false, fa.b, false, fb.b, (short)0, sacc[t], false, false);
            }
        }

        // Online softmax; row n = wave*16 + r + 8*hf lives in vgpr r of each half
        float alpha[8], psum[8];
#pragma unroll
        for (int r = 0; r < 8; ++r) {
            float m = sacc[0][r];
#pragma unroll
            for (int t = 1; t < 4; ++t) m = fmaxf(m, sacc[t][r]);
#pragma unroll
            for (int off = 8; off >= 1; off >>= 1)
                m = fmaxf(m, __shfl_xor(m, off, 16));
            const float Mn = fmaxf(Mrow[r], m * scale);
            alpha[r] = __expf(Mrow[r] - Mn);
            Mrow[r]  = Mn;
            psum[r]  = 0.f;
        }
#pragma unroll
        for (int t = 0; t < 4; ++t) {
#pragma unroll
            for (int r = 0; r < 8; ++r) {
                float p = __expf(sacc[t][r] * scale - Mrow[r]);
                psum[r] += p;
                ps[wave * 16 + r + 8 * hf][t * 16 + l16] = f2bf(p);
            }
        }
#pragma unroll
        for (int r = 0; r < 8; ++r) {
            float s = psum[r];
#pragma unroll
            for (int off = 8; off >= 1; off >>= 1)
                s += __shfl_xor(s, off, 16);
            Lrow[r] = Lrow[r] * alpha[r] + s;
        }
#pragma unroll
        for (int j = 0; j < 8; ++j)
#pragma unroll
            for (int r = 0; r < 8; ++r)
                oacc[j][r] = oacc[j][r] * alpha[r];

        __syncthreads();   // ps visible (and all waves aligned) before P*V^T

        // O += P @ V^T : 16n x 128d, K = m (64)
#pragma unroll
        for (int kk = 0; kk < 2; ++kk) {
            FragBF fa;
            fa.u[0] = *(const ushort8*)&ps[wave * 16 + l16][kk * 32 + hf * 8];
            fa.u[1] = *(const ushort8*)&ps[wave * 16 + l16][kk * 32 + 16 + hf * 8];
#pragma unroll
            for (int j = 0; j < 8; ++j) {
                FragBF fb;
                fb.u[0] = *(const ushort8*)&vs[j * 16 + l16][kk * 32 + hf * 8];
                fb.u[1] = *(const ushort8*)&vs[j * 16 + l16][kk * 32 + 16 + hf * 8];
                oacc[j] = __builtin_amdgcn_wmma_f32_16x16x32_bf16(
                    false, fa.b, false, fb.b, (short)0, oacc[j], false, false);
            }
        }
        __syncthreads();   // before K/V tile overwrite
    }

    // Finalize: out[(head*128+d), n] = O[n][d] / L[n]
#pragma unroll
    for (int r = 0; r < 8; ++r) {
        const float inv = 1.f / Lrow[r];
        const int n = nbase + wave * 16 + r + 8 * hf;
#pragma unroll
        for (int j = 0; j < 8; ++j) {
            const int dd = j * 16 + l16;
            out[(size_t)(head * DHEAD + dd) * N_TOK + n] = f2bf(oacc[j][r] * inv);
        }
    }
}

// ---------------- Launch ----------------
extern "C" void kernel_launch(void* const* d_in, const int* in_sizes, int n_in,
                              void* d_out, int out_size, void* d_ws, size_t ws_size,
                              hipStream_t stream) {
    (void)in_sizes; (void)n_in; (void)out_size; (void)ws_size;
    const float* x      = (const float*)d_in[0];
    const float* norm_w = (const float*)d_in[1];
    const float* norm_b = (const float*)d_in[2];
    const float* qkv_w  = (const float*)d_in[3];
    const float* qkv_b  = (const float*)d_in[4];
    const float* proj_w = (const float*)d_in[5];
    const float* proj_b = (const float*)d_in[6];
    float* out = (float*)d_out;

    char* ws = (char*)d_ws;
    size_t off = 0;
    float* stats = (float*)(ws + off);               off += 256;
    unsigned short* h_bf    = (unsigned short*)(ws + off); off += (size_t)C_DIM * N_TOK * 2;
    unsigned short* qkvw_bf = (unsigned short*)(ws + off); off += (size_t)3 * C_DIM * C_DIM * 2;
    unsigned short* projw_bf= (unsigned short*)(ws + off); off += (size_t)C_DIM * C_DIM * 2;
    unsigned short* qkv_bf  = (unsigned short*)(ws + off); off += (size_t)3 * C_DIM * N_TOK * 2;
    unsigned short* ao_bf   = (unsigned short*)(ws + off); off += (size_t)C_DIM * N_TOK * 2;

    gn_stats<<<NGROUPS, 256, 0, stream>>>(x, stats);
    gn_apply<<<(C_DIM * N_TOK) / (256 * 4), 256, 0, stream>>>(x, stats, norm_w, norm_b, h_bf);
    cvt_w<<<(3 * C_DIM * C_DIM + 255) / 256, 256, 0, stream>>>(qkv_w, proj_w, qkvw_bf, projw_bf);
    gemm_wmma<0><<<dim3(N_TOK / 128, 3 * C_DIM / 128), 256, 0, stream>>>(
        qkvw_bf, h_bf, qkv_b, nullptr, qkv_bf, nullptr);
    flash_attn<<<dim3(N_TOK / 64, NHEAD), 128, 0, stream>>>(qkv_bf, ao_bf);
    gemm_wmma<1><<<dim3(N_TOK / 128, C_DIM / 128), 256, 0, stream>>>(
        projw_bf, ao_bf, proj_b, x, nullptr, out);
}
